// AttentionFlowLayer_70489003262127
// MI455X (gfx1250) — matrix operand order repaired and checked by reading, not compiled, not run on hardware
//
#include <hip/hip_runtime.h>

// ---------------------------------------------------------------------------
// BiDAF attention-flow layer for MI455X (gfx1250), wave32 + WMMA f16->f32.
// B=32, LH=1024, LU=128, H=256.  All GEMMs via v_wmma_f32_16x16x32_f16.
// ---------------------------------------------------------------------------

#define DEV __device__ __forceinline__

typedef _Float16 f16;
typedef __attribute__((ext_vector_type(16))) _Float16 v16h;
typedef __attribute__((ext_vector_type(2)))  _Float16 v2h;
typedef __attribute__((ext_vector_type(8)))  float    v8f;

constexpr int NB = 32;     // batch
constexpr int LH = 1024;   // h sequence length
constexpr int LU = 128;    // u sequence length
constexpr int H  = 256;    // hidden
constexpr int TI = 64;     // h-rows per workgroup tile
constexpr float NEGF = -1e30f;

// ---- workspace layout (bytes); total ~27.3 MB (fits in 192 MB L2) ----
constexpr size_t WS_S    = 0;                                   // f32 [NB][LH][LU]
constexpr size_t WS_A16  = WS_S    + (size_t)NB*LH*LU*4;        // f16 [NB][LH][LU]
constexpr size_t WS_C16  = WS_A16  + (size_t)NB*LH*LU*2;        // f16 [NB][LU][H]
constexpr size_t WS_CMAX = WS_C16  + (size_t)NB*LU*H*2;         // f32 [NB][LU]
constexpr size_t WS_CSUM = WS_CMAX + (size_t)NB*LU*4;           // f32 [NB][LU]

// ---------------------------------------------------------------------------
// WMMA helpers (wave32).  A: 16x32 f16, B: 32x16 f16, C/D: 16x16 f32 (8 VGPR).
// C/D layout per CDNA5 ISA §7.12.2.
// ---------------------------------------------------------------------------
DEV v8f wmma_f16(v16h a, v16h b, v8f c) {
  return __builtin_amdgcn_wmma_f32_16x16x32_f16(
      /*neg_a=*/false, a, /*neg_b=*/false, b,
      /*c_mod=*/(short)0, c, /*reuse_a=*/false, /*reuse_b=*/false);
}

// A fragment from row-major M x K storage (ld = row stride in elems).
// A frag k order used here (ISA table differs): lane m=lane&15, half hf=lane>>4; VGPR r holds K pair
// kk = 16*hf + 2*r.  Pairs contiguous along K -> 32-bit LDS load.
DEV v16h load_a_frag(const f16* p, int ld, int row0, int k0, int lane) {
  const int m  = lane & 15;
  const int hf = lane >> 4;
  const f16* base = p + (size_t)(row0 + m) * ld + k0;
  v16h a;
#pragma unroll
  for (int r = 0; r < 8; ++r) {
    const int kk = 16*hf + 2*r;
    v2h pr = *(const v2h*)(base + kk);
    a[2*r]   = pr[0];
    a[2*r+1] = pr[1];
  }
  return a;
}

// A fragment when the source is K-major: element (m,k) at p[(k0+k)*ld + row0+m].
DEV v16h load_a_frag_kmajor(const f16* p, int ld, int row0, int k0, int lane) {
  const int m  = lane & 15;
  const int hf = lane >> 4;
  const f16* base = p + row0 + m;
  v16h a;
#pragma unroll
  for (int r = 0; r < 8; ++r) {
    const int kk = 16*hf + 2*r;
    a[2*r]   = base[(size_t)(k0 + kk)     * ld];
    a[2*r+1] = base[(size_t)(k0 + kk + 1) * ld];
  }
  return a;
}

// B fragment (32x16, K x N).  Lane n=lane&15, hf=lane>>4; VGPR r holds
// K pair kk = 16*hf + 2*r (cf. sparse-B layout in CDNA5 ISA).
// Variant 1: K contiguous in memory (source is N-major / "transposed"):
// element (k,n) at p[(col0+n)*ld + k0 + k].
DEV v16h load_b_frag_kcontig(const f16* p, int ld, int col0, int k0, int lane) {
  const int n  = lane & 15;
  const int hf = lane >> 4;
  const f16* base = p + (size_t)(col0 + n) * ld + k0 + 16*hf;
  v16h bf;
#pragma unroll
  for (int r = 0; r < 8; ++r) {
    v2h pr = *(const v2h*)(base + 2*r);
    bf[2*r]   = pr[0];
    bf[2*r+1] = pr[1];
  }
  return bf;
}

// Variant 2: row-major K x N source: element (k,n) at p[(k0+k)*ld + col0+n].
DEV v16h load_b_frag_rowmajor(const f16* p, int ld, int col0, int k0, int lane) {
  const int n  = lane & 15;
  const int hf = lane >> 4;
  const f16* base = p + (size_t)(k0 + 16*hf) * ld + col0 + n;
  v16h bf;
#pragma unroll
  for (int r = 0; r < 8; ++r) {
    bf[2*r]   = base[(size_t)(2*r)     * ld];
    bf[2*r+1] = base[(size_t)(2*r + 1) * ld];
  }
  return bf;
}

// ---------------------------------------------------------------------------
// K1: per (batch, 64-row h-tile):  S -> row softmax -> a_t -> U_tilde,
//     writes out[:, :, 0:3H), and S (f32) + a_t (f16) to workspace.
// ---------------------------------------------------------------------------
__global__ __launch_bounds__(256)
void k1_score_softmax_ut(const float* __restrict__ h, const float* __restrict__ u,
                         const unsigned char* __restrict__ umask,
                         const float* __restrict__ w, const float* __restrict__ bptr,
                         float* __restrict__ Sws, f16* __restrict__ a16ws,
                         float* __restrict__ out)
{
  __shared__ f16   sh_u16[LU * H];    // 64 KB: u as f16, row-major [j][k]
  __shared__ f16   sh_a16[TI * LU];   // 16 KB: a_t tile f16
  __shared__ float sh_scr[TI * H];    // 64 KB: {hA f16 | S f32 tile} then U_tilde f32
  __shared__ float sh_w[3 * H];
  __shared__ float sh_hw[TI];
  __shared__ float sh_hu[LU];
  __shared__ float sh_rmax[TI];
  __shared__ float sh_rsum[TI];
  __shared__ float sh_red[256];

  const int tid  = threadIdx.x;
  const int lane = tid & 31;
  const int wv   = tid >> 5;
  const int b    = blockIdx.y;
  const int i0   = blockIdx.x * TI;

  const float* hB = h + (size_t)b * LH * H;
  const float* uB = u + (size_t)b * LU * H;
  const float bias = bptr[0];

  for (int t = tid; t < 3 * H; t += 256) sh_w[t] = w[t];
  __syncthreads();

  // ---- stage u as f16 (vectorized) ----
  {
    const float4* u4 = (const float4*)uB;
    for (int idx = tid; idx < LU * H / 4; idx += 256) {
      float4 v = u4[idx];
      int o = idx * 4;
      sh_u16[o+0] = (f16)v.x; sh_u16[o+1] = (f16)v.y;
      sh_u16[o+2] = (f16)v.z; sh_u16[o+3] = (f16)v.w;
    }
  }
  // ---- hA = (h tile ∘ w_hu) as f16 in first half of sh_scr ----
  {
    f16* hA = (f16*)sh_scr;
    const float* whu = sh_w + 2 * H;
    for (int idx = tid; idx < TI * H / 4; idx += 256) {
      int i = idx >> 6;               // H/4 = 64 float4 per row
      int c = (idx & 63) * 4;
      float4 v = *(const float4*)(hB + (size_t)(i0 + i) * H + c);
      int o = i * H + c;
      hA[o+0] = (f16)(v.x * whu[c+0]);
      hA[o+1] = (f16)(v.y * whu[c+1]);
      hA[o+2] = (f16)(v.z * whu[c+2]);
      hA[o+3] = (f16)(v.w * whu[c+3]);
    }
  }
  // ---- hw[i] = h[i]·w_h, hu[j] = u[j]·w_u ----
  if (tid < TI) {
    float s = 0.f;
    const float* row = hB + (size_t)(i0 + tid) * H;
    for (int k = 0; k < H; ++k) s += row[k] * sh_w[k];
    sh_hw[tid] = s;
  } else if (tid < TI + LU) {
    const int j = tid - TI;
    float s = 0.f;
    const float* row = uB + (size_t)j * H;
    for (int k = 0; k < H; ++k) s += row[k] * sh_w[H + k];
    sh_hu[j] = s;
  }
  __syncthreads();

  // ---- S tile = hA(64x256) @ u^T(256x128) via WMMA ----
  {
    const f16* hA = (const f16*)sh_scr;
    float* Stile = sh_scr + (TI * H / 2);    // second 32 KB, f32 [TI][LU]
    const int rb = wv >> 1;                  // row block 0..3 (16 rows each)
    const int c0 = (wv & 1) * 64;            // 64 cols per wave
    v8f acc[4] = {};
    for (int k0 = 0; k0 < H; k0 += 32) {
      v16h a = load_a_frag(hA, H, rb * 16, k0, lane);
#pragma unroll
      for (int nt = 0; nt < 4; ++nt) {
        v16h bf = load_b_frag_kcontig(sh_u16, H, c0 + nt * 16, k0, lane);
        acc[nt] = wmma_f16(a, bf, acc[nt]);
      }
    }
    const int nn = lane & 15;
    const int hf = lane >> 4;
#pragma unroll
    for (int nt = 0; nt < 4; ++nt) {
#pragma unroll
      for (int v = 0; v < 8; ++v) {
        const int row = rb * 16 + v + 8 * hf;
        const int col = c0 + nt * 16 + nn;
        float s = acc[nt][v] + sh_hw[row] + sh_hu[col] + bias;
        Sws[((size_t)(b * LH + i0 + row)) * LU + col] = s;   // raw S for K2/K4
        Stile[row * LU + col] = umask[b * LU + col] ? s : NEGF;
      }
    }
  }
  __syncthreads();

  // ---- row softmax over j (4 threads per row, 32 cols each) ----
  {
    float* Stile = sh_scr + (TI * H / 2);
    const int row = tid >> 2;
    const int q   = tid & 3;
    float mx = NEGF;
    for (int c = q * 32; c < q * 32 + 32; ++c) mx = fmaxf(mx, Stile[row * LU + c]);
    sh_red[tid] = mx;
    __syncthreads();
    if (q == 0)
      sh_rmax[row] = fmaxf(fmaxf(sh_red[tid], sh_red[tid+1]),
                           fmaxf(sh_red[tid+2], sh_red[tid+3]));
    __syncthreads();
    const float rm = sh_rmax[row];
    float sm = 0.f;
    for (int c = q * 32; c < q * 32 + 32; ++c) {
      float e = __expf(Stile[row * LU + c] - rm);
      Stile[row * LU + c] = e;
      sm += e;
    }
    sh_red[tid] = sm;
    __syncthreads();
    if (q == 0) sh_rsum[row] = sh_red[tid] + sh_red[tid+1] + sh_red[tid+2] + sh_red[tid+3];
    __syncthreads();
    const float inv = 1.f / sh_rsum[row];
    for (int c = q * 32; c < q * 32 + 32; ++c) {
      float a = Stile[row * LU + c] * inv;
      sh_a16[row * LU + c] = (f16)a;
      a16ws[((size_t)(b * LH + i0 + row)) * LU + c] = (f16)a;
    }
  }
  __syncthreads();

  // ---- U_tilde = a(64x128) @ u(128x256) via WMMA; stash f32 in sh_scr ----
  {
    const int mt = wv >> 1;             // 0..3
    const int n0 = (wv & 1) * 128;      // 128 cols per wave
    v8f acc[8] = {};
    for (int k0 = 0; k0 < LU; k0 += 32) {
      v16h a = load_a_frag(sh_a16, LU, mt * 16, k0, lane);
#pragma unroll
      for (int nt = 0; nt < 8; ++nt) {
        v16h bf = load_b_frag_rowmajor(sh_u16, H, n0 + nt * 16, k0, lane);
        acc[nt] = wmma_f16(a, bf, acc[nt]);
      }
    }
    float* Ut = sh_scr;                 // Stile/hA dead now
    const int nn = lane & 15;
    const int hf = lane >> 4;
#pragma unroll
    for (int nt = 0; nt < 8; ++nt)
#pragma unroll
      for (int v = 0; v < 8; ++v)
        Ut[(mt * 16 + v + 8 * hf) * H + n0 + nt * 16 + nn] = acc[nt][v];
  }
  __syncthreads();

  // ---- write out cols [0,3H): h | U_tilde | h∘U_tilde (float4 stores) ----
  {
    const float* Ut = sh_scr;
    float* outB = out + ((size_t)b * LH + i0) * (4 * H);
    for (int idx = tid; idx < TI * H / 4; idx += 256) {
      int i = idx >> 6;
      int c = (idx & 63) * 4;
      float4 hv = *(const float4*)(hB + (size_t)(i0 + i) * H + c);
      float4 ut = *(const float4*)(Ut + i * H + c);
      float* o = outB + (size_t)i * (4 * H);
      *(float4*)(o + c)         = hv;
      *(float4*)(o + H + c)     = ut;
      float4 p; p.x = hv.x*ut.x; p.y = hv.y*ut.y; p.z = hv.z*ut.z; p.w = hv.w*ut.w;
      *(float4*)(o + 2*H + c)   = p;
    }
  }
}

// ---------------------------------------------------------------------------
// K2: masked column max/sum of S per batch (for the i-axis softmax b_t).
// ---------------------------------------------------------------------------
__global__ __launch_bounds__(256)
void k2_colstats(const float* __restrict__ Sws, const unsigned char* __restrict__ hmask,
                 float* __restrict__ cmax, float* __restrict__ csum)
{
  __shared__ float red[256];
  __shared__ float cm_sh[LU];
  const int b    = blockIdx.x;
  const int tid  = threadIdx.x;
  const int j    = tid & (LU - 1);
  const int half = tid >> 7;
  const float* Sb = Sws + (size_t)b * LH * LU;
  const unsigned char* mb = hmask + (size_t)b * LH;

  float mx = NEGF;
  for (int i = half * (LH / 2); i < (half + 1) * (LH / 2); ++i)
    if (mb[i]) mx = fmaxf(mx, Sb[(size_t)i * LU + j]);
  red[tid] = mx;
  __syncthreads();
  if (half == 0) cm_sh[j] = fmaxf(red[j], red[j + LU]);
  __syncthreads();
  const float cm = cm_sh[j];
  float sm = 0.f;
  for (int i = half * (LH / 2); i < (half + 1) * (LH / 2); ++i)
    if (mb[i]) sm += __expf(Sb[(size_t)i * LU + j] - cm);
  red[tid] = sm;
  __syncthreads();
  if (half == 0) {
    cmax[(size_t)b * LU + j] = cm;
    csum[(size_t)b * LU + j] = red[j] + red[j + LU];
  }
}

// ---------------------------------------------------------------------------
// K3: C = a_t^T(128x1024) @ h(1024x256) per batch; block = (N-half, batch).
// ---------------------------------------------------------------------------
__global__ __launch_bounds__(256)
void k3_context(const f16* __restrict__ a16ws, const float* __restrict__ h,
                f16* __restrict__ C16)
{
  __shared__ f16 aPan[32 * LU];    // 8 KB [kk][j]   (K-major for A)
  __shared__ f16 hPan[32 * 128];   // 8 KB [kk][c]
  const int tid = threadIdx.x, lane = tid & 31, wv = tid >> 5;
  const int b   = blockIdx.y;
  const int n0  = blockIdx.x * 128;
  const float* hB = h + (size_t)b * LH * H;
  const f16*   aB = a16ws + (size_t)b * LH * LU;

  v8f acc[8] = {};
  for (int k0 = 0; k0 < LH; k0 += 32) {
    if (k0 + 32 < LH)
      __builtin_prefetch(hB + (size_t)(k0 + 32) * H + n0, 0, 0);  // global_prefetch
    for (int idx = tid; idx < 32 * LU; idx += 256)
      aPan[idx] = aB[(size_t)(k0 + (idx >> 7)) * LU + (idx & 127)];
    for (int idx = tid; idx < 32 * 128; idx += 256)
      hPan[idx] = (f16)hB[(size_t)(k0 + (idx >> 7)) * H + n0 + (idx & 127)];
    __syncthreads();
    v16h a = load_a_frag_kmajor(aPan, LU, wv * 16, 0, lane);
#pragma unroll
    for (int nt = 0; nt < 8; ++nt) {
      v16h bf = load_b_frag_rowmajor(hPan, 128, nt * 16, 0, lane);
      acc[nt] = wmma_f16(a, bf, acc[nt]);
    }
    __syncthreads();
  }
  const int nn = lane & 15, hf = lane >> 4;
#pragma unroll
  for (int nt = 0; nt < 8; ++nt)
#pragma unroll
    for (int v = 0; v < 8; ++v) {
      const int j = wv * 16 + v + 8 * hf;
      C16[((size_t)(b * LU + j)) * H + n0 + nt * 16 + nn] = (f16)acc[nt][v];
    }
}

// ---------------------------------------------------------------------------
// K4: b_t tile from S + col-stats; H_tilde = b_t(64x128) @ C(128x256);
//     writes out[:, :, 3H:4H) = h ∘ H_tilde.
// ---------------------------------------------------------------------------
__global__ __launch_bounds__(256)
void k4_htilde(const float* __restrict__ h, const unsigned char* __restrict__ hmask,
               const float* __restrict__ Sws, const f16* __restrict__ C16,
               const float* __restrict__ cmax, const float* __restrict__ csum,
               float* __restrict__ out)
{
  __shared__ f16   sh_c16[LU * H];   // 64 KB
  __shared__ f16   sh_bt[TI * LU];   // 16 KB
  __shared__ float sh_ht[TI * H];    // 64 KB
  __shared__ float sh_cm[LU];
  __shared__ float sh_cs[LU];

  const int tid = threadIdx.x, lane = tid & 31, wv = tid >> 5;
  const int b   = blockIdx.y;
  const int i0  = blockIdx.x * TI;
  const float* hB = h + (size_t)b * LH * H;

  for (int idx = tid; idx < LU * H; idx += 256)
    sh_c16[idx] = C16[(size_t)b * LU * H + idx];
  if (tid < LU) {
    sh_cm[tid] = cmax[(size_t)b * LU + tid];
    sh_cs[tid] = csum[(size_t)b * LU + tid];
  }
  __syncthreads();

  for (int idx = tid; idx < TI * LU; idx += 256) {
    const int i = idx >> 7, j = idx & 127;
    const int gi = i0 + i;
    float v = 0.f;
    if (hmask[(size_t)b * LH + gi])
      v = __expf(Sws[((size_t)(b * LH + gi)) * LU + j] - sh_cm[j]) / sh_cs[j];
    sh_bt[idx] = (f16)v;
  }
  __syncthreads();

  {
    const int mt = wv >> 1;
    const int n0 = (wv & 1) * 128;
    v8f acc[8] = {};
    for (int k0 = 0; k0 < LU; k0 += 32) {
      v16h a = load_a_frag(sh_bt, LU, mt * 16, k0, lane);
#pragma unroll
      for (int nt = 0; nt < 8; ++nt) {
        v16h bf = load_b_frag_rowmajor(sh_c16, H, n0 + nt * 16, k0, lane);
        acc[nt] = wmma_f16(a, bf, acc[nt]);
      }
    }
    const int nn = lane & 15, hf = lane >> 4;
#pragma unroll
    for (int nt = 0; nt < 8; ++nt)
#pragma unroll
      for (int v = 0; v < 8; ++v)
        sh_ht[(mt * 16 + v + 8 * hf) * H + n0 + nt * 16 + nn] = acc[nt][v];
  }
  __syncthreads();

  float* outB = out + ((size_t)b * LH + i0) * (4 * H);
  for (int idx = tid; idx < TI * H / 4; idx += 256) {
    const int i = idx >> 6;
    const int c = (idx & 63) * 4;
    float4 hv = *(const float4*)(hB + (size_t)(i0 + i) * H + c);
    float4 ht = *(const float4*)(sh_ht + i * H + c);
    float4 p; p.x = hv.x*ht.x; p.y = hv.y*ht.y; p.z = hv.z*ht.z; p.w = hv.w*ht.w;
    *(float4*)(outB + (size_t)i * (4 * H) + 3 * H + c) = p;
  }
}

// ---------------------------------------------------------------------------
extern "C" void kernel_launch(void* const* d_in, const int* in_sizes, int n_in,
                              void* d_out, int out_size, void* d_ws, size_t ws_size,
                              hipStream_t stream)
{
  (void)in_sizes; (void)n_in; (void)out_size; (void)ws_size;
  const float*         h     = (const float*)d_in[0];
  const float*         u     = (const float*)d_in[1];
  const unsigned char* hmask = (const unsigned char*)d_in[2];  // jnp bool = 1 byte
  const unsigned char* umask = (const unsigned char*)d_in[3];
  const float*         w     = (const float*)d_in[4];
  const float*         bias  = (const float*)d_in[5];
  float* out = (float*)d_out;
  char*  ws  = (char*)d_ws;

  float* Sws  = (float*)(ws + WS_S);
  f16*   a16  = (f16*)(ws + WS_A16);
  f16*   C16  = (f16*)(ws + WS_C16);
  float* cmax = (float*)(ws + WS_CMAX);
  float* csum = (float*)(ws + WS_CSUM);

  k1_score_softmax_ut<<<dim3(LH / TI, NB), 256, 0, stream>>>(h, u, umask, w, bias,
                                                             Sws, a16, out);
  k2_colstats<<<dim3(NB), 256, 0, stream>>>(Sws, hmask, cmax, csum);
  k3_context<<<dim3(2, NB), 256, 0, stream>>>(a16, h, C16);
  k4_htilde<<<dim3(LH / TI, NB), 256, 0, stream>>>(h, hmask, Sws, C16, cmax, csum, out);
}
